// ClothesBasedAdversarialLossWithMemoryBank_27212912788233
// MI455X (gfx1250) — matrix operand order repaired and checked by reading, not compiled, not run on hardware
//
#include <hip/hip_runtime.h>
#include <hip/hip_bf16.h>
#include <math.h>

// ---------------- problem constants ----------------
#define NUM_CLOTHES 50000
#define FEAT        256
#define BATCH       1024
#define NTILE       3125        // 50000 / 16
#define NS          32          // column slices (== wave32 lanes, used in finalize)
#define CAP         2048        // per-row positive-logit capacity (~500 expected)
#define SCALE_F     16.0f
#define EPS_F       0.1f

typedef __bf16 bf16_t;
typedef __attribute__((ext_vector_type(16))) __bf16 v16bf;
typedef __attribute__((ext_vector_type(8)))  __bf16 v8bf;
typedef __attribute__((ext_vector_type(8)))  float  v8f;

// ---------------- 1) zero workspace region ----------------
__global__ __launch_bounds__(256)
void zero_kernel(float* __restrict__ p, long long n) {
    long long i = (long long)blockIdx.x * blockDim.x + threadIdx.x;
    long long stride = (long long)gridDim.x * blockDim.x;
    for (; i < n; i += stride) p[i] = 0.0f;
}

// ---------------- 2) scatter-sum of inputs per target class ----------------
__global__ __launch_bounds__(256)
void scatter_kernel(const float* __restrict__ inputs, const int* __restrict__ targets,
                    float* __restrict__ sums, float* __restrict__ counts) {
    int idx = blockIdx.x * blockDim.x + threadIdx.x;   // BATCH*FEAT threads
    int b = idx >> 8;                                  // FEAT == 256
    int d = idx & 255;
    int t = targets[b];
    atomicAdd(&sums[t * FEAT + d], inputs[idx]);
    if (d == 0) atomicAdd(&counts[t], 1.0f);
}

// ---------------- 3) memory bank: mean-or-keep, L2 normalize, -> bf16 ----------------
__global__ __launch_bounds__(256)
void mem_norm_kernel(const float* __restrict__ sums, const float* __restrict__ counts,
                     const float* __restrict__ memory, bf16_t* __restrict__ memN) {
    int wid  = threadIdx.x >> 5;
    int lane = threadIdx.x & 31;
    int c = blockIdx.x * 8 + wid;
    if (c >= NUM_CLOTHES) return;
    float cnt = counts[c];
    const float* src = (cnt > 0.0f) ? (sums + (size_t)c * FEAT) : (memory + (size_t)c * FEAT);
    float inv = (cnt > 0.0f) ? (1.0f / cnt) : 1.0f;
    float v[8]; float ss = 0.0f;
#pragma unroll
    for (int i = 0; i < 8; ++i) { v[i] = src[lane + i * 32] * inv; ss += v[i] * v[i]; }
#pragma unroll
    for (int d = 1; d < 32; d <<= 1) ss += __shfl_xor(ss, d, 32);
    float r = 1.0f / fmaxf(sqrtf(ss), 1e-12f);
#pragma unroll
    for (int i = 0; i < 8; ++i)
        memN[(size_t)c * FEAT + lane + i * 32] = (bf16_t)(v[i] * r);
}

// ---------------- 4) input rows: L2 normalize -> bf16 ----------------
__global__ __launch_bounds__(256)
void in_norm_kernel(const float* __restrict__ inputs, bf16_t* __restrict__ inN) {
    int wid  = threadIdx.x >> 5;
    int lane = threadIdx.x & 31;
    int b = blockIdx.x * 8 + wid;
    if (b >= BATCH) return;
    const float* src = inputs + (size_t)b * FEAT;
    float v[8]; float ss = 0.0f;
#pragma unroll
    for (int i = 0; i < 8; ++i) { v[i] = src[lane + i * 32]; ss += v[i] * v[i]; }
#pragma unroll
    for (int d = 1; d < 32; d <<= 1) ss += __shfl_xor(ss, d, 32);
    float r = 1.0f / fmaxf(sqrtf(ss), 1e-12f);
#pragma unroll
    for (int i = 0; i < 8; ++i)
        inN[(size_t)b * FEAT + lane + i * 32] = (bf16_t)(v[i] * r);
}

// ---------------- 5) WMMA GEMM + fused logsumexp epilogue ----------------
// grid = (64 row-bands of 16, NS column slices), block = 32 (one wave).
// A fragments (16 rows x 256 K) kept in registers across the whole column loop.
__global__ __launch_bounds__(32)
void gemm_pass1_kernel(const bf16_t* __restrict__ inN, const bf16_t* __restrict__ memN,
                       const float* __restrict__ posmask, const int* __restrict__ targets,
                       float* __restrict__ negsum_part, float* __restrict__ poscnt_part,
                       int* __restrict__ rowfill, float* __restrict__ posvals,
                       float* __restrict__ s_ident) {
    const int lane = threadIdx.x & 31;
    const int half = lane >> 4;     // 0: lanes 0-15, 1: lanes 16-31
    const int lm   = lane & 15;
    const int m0   = blockIdx.x * 16;
    const int slice = blockIdx.y;

    // --- A fragments (contiguous K order, same as B: half 0 -> K 0..15, half 1 -> K 16..31) ---
    v16bf afrag[8];
    {
        const bf16_t* arow = inN + (size_t)(m0 + lm) * FEAT;
        const int base = half * 16;
#pragma unroll
        for (int k8 = 0; k8 < 8; ++k8) {
            const int kk = k8 * 32;
            v8bf alo = *(const v8bf*)(arow + kk + base);        // K = base+0..7
            v8bf ahi = *(const v8bf*)(arow + kk + base + 8);    // K = base+8..15
            afrag[k8] = __builtin_shufflevector(alo, ahi,
                0, 1, 2, 3, 4, 5, 6, 7, 8, 9, 10, 11, 12, 13, 14, 15);
        }
    }

    // targets for the 8 rows this lane's accumulator slots map to (M = half*8 + i)
    int tr[8];
#pragma unroll
    for (int i = 0; i < 8; ++i) tr[i] = targets[m0 + half * 8 + i];

    float negp[8], pcnt[8];
#pragma unroll
    for (int i = 0; i < 8; ++i) { negp[i] = 0.0f; pcnt[i] = 0.0f; }

    for (int t = slice; t < NTILE; t += NS) {
        const int n0  = t * 16;
        const int col = n0 + lm;                 // this lane's N (B and C/D layouts)
        // --- B fragment base: lane covers K = half*16 .. half*16+15 of each 32-window ---
        const bf16_t* brow = memN + (size_t)col * FEAT + half * 16;
        v8f acc = {0.f, 0.f, 0.f, 0.f, 0.f, 0.f, 0.f, 0.f};
#pragma unroll
        for (int k8 = 0; k8 < 8; ++k8) {
            v16bf bfrag = *(const v16bf*)(brow + k8 * 32);
            acc = __builtin_amdgcn_wmma_f32_16x16x32_bf16(
                false, afrag[k8], false, bfrag, (short)0, acc, false, false);
        }
        // --- fused epilogue: C layout VGPR i -> row m0 + half*8 + i, col = n0 + lm ---
#pragma unroll
        for (int i = 0; i < 8; ++i) {
            const int   m = m0 + half * 8 + i;
            const float s = acc[i] * SCALE_F;
            const float p = posmask[m * NUM_CLOTHES + col];
            negp[i] += expf(s) * (1.0f - p);
            pcnt[i] += p;
            if (p > 0.0f) {
                int slot = atomicAdd(&rowfill[m], 1);
                if (slot < CAP) posvals[m * CAP + slot] = s;
            }
            if (col == tr[i]) s_ident[m] = s;
        }
    }

    // --- reduce across the 16 lanes of each half (same rows, different N) ---
#pragma unroll
    for (int i = 0; i < 8; ++i) {
        float v = negp[i], q = pcnt[i];
#pragma unroll
        for (int d = 1; d < 16; d <<= 1) {
            v += __shfl_xor(v, d, 32);
            q += __shfl_xor(q, d, 32);
        }
        if (lm == 0) {
            const int m = m0 + half * 8 + i;
            negsum_part[slice * BATCH + m] = v;
            poscnt_part[slice * BATCH + m] = q;
        }
    }
}

// ---------------- 6) per-row loss (wave per row; NS==32 partials map to lanes) ----------------
__global__ __launch_bounds__(256)
void finalize_rows_kernel(const float* __restrict__ negsum_part, const float* __restrict__ poscnt_part,
                          const int* __restrict__ rowfill, const float* __restrict__ posvals,
                          const float* __restrict__ s_ident, float* __restrict__ rowloss) {
    int wid  = threadIdx.x >> 5;
    int lane = threadIdx.x & 31;
    int row = blockIdx.x * 8 + wid;
    if (row >= BATCH) return;
    float neg = negsum_part[lane * BATCH + row];
    float P   = poscnt_part[lane * BATCH + row];
#pragma unroll
    for (int d = 1; d < 32; d <<= 1) {
        neg += __shfl_xor(neg, d, 32);
        P   += __shfl_xor(P, d, 32);
    }
    int K = rowfill[row]; if (K > CAP) K = CAP;
    float acc = 0.0f;
    for (int k = lane; k < K; k += 32) {
        float sv = posvals[row * CAP + k];
        acc += logf(neg + expf(sv)) - sv;
    }
#pragma unroll
    for (int d = 1; d < 32; d <<= 1) acc += __shfl_xor(acc, d, 32);
    if (lane == 0) {
        float st = s_ident[row];
        rowloss[row] = (1.0f - EPS_F) * (logf(neg + expf(st)) - st) + (EPS_F / P) * acc;
    }
}

// ---------------- 7) deterministic mean over rows ----------------
__global__ __launch_bounds__(256)
void final_reduce_kernel(const float* __restrict__ rowloss, float* __restrict__ out) {
    __shared__ float sm[256];
    float a = 0.0f;
    for (int i = threadIdx.x; i < BATCH; i += 256) a += rowloss[i];
    sm[threadIdx.x] = a;
    __syncthreads();
    for (int s = 128; s > 0; s >>= 1) {
        if (threadIdx.x < s) sm[threadIdx.x] += sm[threadIdx.x + s];
        __syncthreads();
    }
    if (threadIdx.x == 0) out[0] = sm[0] / (float)BATCH;
}

// ---------------- host launch ----------------
extern "C" void kernel_launch(void* const* d_in, const int* in_sizes, int n_in,
                              void* d_out, int out_size, void* d_ws, size_t ws_size,
                              hipStream_t stream) {
    (void)in_sizes; (void)n_in; (void)out_size; (void)ws_size;
    const float* inputs  = (const float*)d_in[0];   // [1024,256]
    const float* memory  = (const float*)d_in[1];   // [50000,256]
    const float* posmask = (const float*)d_in[2];   // [1024,50000]
    const int*   targets = (const int*)d_in[3];     // [1024]
    float* out = (float*)d_out;

    // workspace layout (256B aligned)
    char* ws = (char*)d_ws;
    size_t off = 0;
    auto take = [&](size_t bytes) { size_t o = off; off += (bytes + 255) & ~(size_t)255; return o; };
    float*  sums        = (float*) (ws + take((size_t)NUM_CLOTHES * FEAT * 4)); // zeroed
    float*  counts      = (float*) (ws + take((size_t)NUM_CLOTHES * 4));        // zeroed
    float*  negsum_part = (float*) (ws + take((size_t)NS * BATCH * 4));         // zeroed
    float*  poscnt_part = (float*) (ws + take((size_t)NS * BATCH * 4));         // zeroed
    int*    rowfill     = (int*)   (ws + take((size_t)BATCH * 4));              // zeroed
    size_t zero_bytes = off;                                                    // zero-region end
    bf16_t* memN        = (bf16_t*)(ws + take((size_t)NUM_CLOTHES * FEAT * 2));
    bf16_t* inN         = (bf16_t*)(ws + take((size_t)BATCH * FEAT * 2));
    float*  posvals     = (float*) (ws + take((size_t)BATCH * CAP * 4));
    float*  s_ident     = (float*) (ws + take((size_t)BATCH * 4));
    float*  rowloss     = (float*) (ws + take((size_t)BATCH * 4));

    long long zeroN = (long long)(zero_bytes / 4);
    zero_kernel<<<4096, 256, 0, stream>>>((float*)ws, zeroN);
    scatter_kernel<<<BATCH, 256, 0, stream>>>(inputs, targets, sums, counts);
    mem_norm_kernel<<<(NUM_CLOTHES + 7) / 8, 256, 0, stream>>>(sums, counts, memory, memN);
    in_norm_kernel<<<(BATCH + 7) / 8, 256, 0, stream>>>(inputs, inN);
    gemm_pass1_kernel<<<dim3(BATCH / 16, NS), 32, 0, stream>>>(
        inN, memN, posmask, targets, negsum_part, poscnt_part, rowfill, posvals, s_ident);
    finalize_rows_kernel<<<BATCH / 8, 256, 0, stream>>>(
        negsum_part, poscnt_part, rowfill, posvals, s_ident, rowloss);
    final_reduce_kernel<<<1, 256, 0, stream>>>(rowloss, out);
}
